// MultiHeadAttention_65386582114568
// MI455X (gfx1250) — matrix operand order repaired and checked by reading, not compiled, not run on hardware
//
#include <hip/hip_runtime.h>

// ---------------------------------------------------------------------------
// MultiHeadAttention for MI455X (gfx1250, wave32, WMMA + async LDS copies).
// Pipeline (BF16 matrix inputs, F32 accumulation):
//   0) convert x, states, Wq, Wk, Wv, Wp  f32 -> bf16
//   1) q  = x @ Wq[h]^T          (per b,h)     1024x512x512
//   2) k  = states @ Wk[h]^T     (per b,h)     1024x512x512
//   3) vT = Wv[h] @ states^T     (per b,h)      512x1024x512
//   4) scores = q @ k^T          (per b,h)     1024x1024x512   -> bf16
//   5) attn = softmax(mask(scale*scores))      rows of 1024, f32 in regs
//   6) ctx = attn @ vT^T         (per b,h)     1024x512x1024   -> bf16, ldc=4096
//   7) out = ctx_cat @ Wp^T      (per b)       1024x512x4096   -> f32 d_out
// One generic bf16 "C = A * B^T" WMMA GEMM kernel (double-buffered LDS with
// GLOBAL_LOAD_ASYNC_TO_LDS_B128) serves phases 1-4, 6, 7.
// ---------------------------------------------------------------------------

typedef __attribute__((ext_vector_type(16))) __bf16 v16bf;
typedef __attribute__((ext_vector_type(8)))  __bf16 v8bf;
typedef __attribute__((ext_vector_type(8)))  float  v8f;
typedef int v4i_vs __attribute__((vector_size(16)));   // matches builtin param type

static constexpr int Bsz = 4;
static constexpr int LQn = 1024;
static constexpr int LKn = 1024;
static constexpr int Dn  = 512;
static constexpr int Hn  = 8;
static constexpr int HDn = Hn * Dn;   // 4096

#if defined(__gfx1250__) && __has_builtin(__builtin_amdgcn_global_load_async_to_lds_b128)
#define HAS_ASYNC 1
#else
#define HAS_ASYNC 0
#endif

__device__ __forceinline__ __bf16 f32_to_bf16(float f) {
    unsigned u = __builtin_bit_cast(unsigned, f);
    unsigned r = (u + 0x7FFFu + ((u >> 16) & 1u)) >> 16;
    unsigned short s = (unsigned short)r;
    return __builtin_bit_cast(__bf16, s);
}
__device__ __forceinline__ float bf16_to_f32(__bf16 h) {
    unsigned short s = __builtin_bit_cast(unsigned short, h);
    unsigned u = ((unsigned)s) << 16;
    return __builtin_bit_cast(float, u);
}

template <typename T> __device__ __forceinline__ T cvt_out(float f);
template <> __device__ __forceinline__ float  cvt_out<float>(float f)  { return f; }
template <> __device__ __forceinline__ __bf16 cvt_out<__bf16>(float f) { return f32_to_bf16(f); }

__device__ __forceinline__ v16bf mkfrag(v8bf lo, v8bf hi) {
    v16bf f;
#pragma unroll
    for (int i = 0; i < 8; i++) { f[i] = lo[i]; f[i + 8] = hi[i]; }
    return f;
}

// 16-byte global -> LDS copy; async (ASYNCcnt, bypasses VGPRs) when available.
__device__ __forceinline__ void cp16_g2l(const __bf16* g, __bf16* l) {
#if HAS_ASYNC
    __builtin_amdgcn_global_load_async_to_lds_b128(
        (__attribute__((address_space(1))) v4i_vs*)g,
        (__attribute__((address_space(3))) v4i_vs*)l, 0, 0);
#else
    *(v8bf*)l = *(const v8bf*)g;
#endif
}

__device__ __forceinline__ void wait_async_copies() {
#if HAS_ASYNC
#if __has_builtin(__builtin_amdgcn_s_wait_asynccnt)
    __builtin_amdgcn_s_wait_asynccnt(0);
#else
    asm volatile("s_wait_asynccnt 0x0" ::: "memory");
#endif
#endif
}

// Batch pointer map: offset(z) = (z / div) * sq + (z % div) * sr   (elements)
struct BMap { long long sq; long long sr; int div; };

// ---------------------------------------------------------------------------
// Generic bf16 GEMM:  C[m,n] = sum_k A[m,k] * B[n,k]   (C = A * B^T)
// A: M x K row-major (lda), B: N x K row-major (ldb), C: M x N (ldc).
// WG tile 128x128, 8 waves each computing 64x32 (4x2 WMMA 16x16 tiles).
// K advanced 64 per LDS slab (2x V_WMMA_F32_16X16X32_BF16 depth), slabs
// double-buffered so async HBM->LDS copies overlap the WMMA pipe.
// M, N multiples of 128; K multiple of 64 (true for all call sites here).
// ---------------------------------------------------------------------------
static constexpr int TILE_M = 128;
static constexpr int TILE_N = 128;
static constexpr int TILE_K = 64;
static constexpr int LDSS   = 72;   // padded row stride (bf16): conflict-free b128

template <typename OUT_T>
__global__ __launch_bounds__(256)
void gemm_bf16_abt(const __bf16* __restrict__ Ag, const __bf16* __restrict__ Bg,
                   OUT_T* __restrict__ Cg, int M, int N, int K,
                   int lda, int ldb, int ldc, BMap am, BMap bm, BMap cm)
{
    __shared__ __bf16 As[2][TILE_M * LDSS];   // 2 x 18 KB
    __shared__ __bf16 Bs[2][TILE_N * LDSS];   // 2 x 18 KB

    const int z = blockIdx.z;
    Ag += (long long)(z / am.div) * am.sq + (long long)(z % am.div) * am.sr;
    Bg += (long long)(z / bm.div) * bm.sq + (long long)(z % bm.div) * bm.sr;
    Cg += (long long)(z / cm.div) * cm.sq + (long long)(z % cm.div) * cm.sr;

    const int mBase = blockIdx.y * TILE_M;
    const int nBase = blockIdx.x * TILE_N;

    const int t    = threadIdx.x;
    const int lane = t & 31;
    const int wave = t >> 5;     // 0..7
    const int wm   = wave & 1;   // 2 waves along M
    const int wn   = wave >> 1;  // 4 waves along N
    const int half = lane >> 4;  // 0/1
    const int lr   = lane & 15;

    v8f acc[4][2];
#pragma unroll
    for (int i = 0; i < 4; i++)
#pragma unroll
        for (int j = 0; j < 2; j++)
#pragma unroll
            for (int r = 0; r < 8; r++) acc[i][j][r] = 0.0f;

    // cooperative slab copy: 128 rows x 64 bf16 (128B/row); 8 threads/row x 16B
    const int lrow = t >> 3;        // 0..31
    const int lcb  = (t & 7) * 8;   // element offset 0..56 step 8

    auto issue_slab = [&](int buf, int kt) {
#pragma unroll
        for (int rr = 0; rr < TILE_M; rr += 32) {
            cp16_g2l(&Ag[(size_t)(mBase + rr + lrow) * lda + kt + lcb],
                     &As[buf][(rr + lrow) * LDSS + lcb]);
            cp16_g2l(&Bg[(size_t)(nBase + rr + lrow) * ldb + kt + lcb],
                     &Bs[buf][(rr + lrow) * LDSS + lcb]);
        }
    };

    const int nSlab = K / TILE_K;
    issue_slab(0, 0);
    wait_async_copies();
    __syncthreads();

    for (int it = 0; it < nSlab; ++it) {
        const int cur = it & 1;
        if (it + 1 < nSlab) issue_slab(cur ^ 1, (it + 1) * TILE_K);

#pragma unroll
        for (int kk = 0; kk < TILE_K; kk += 32) {
            // A frag (16-bit A 16x32): lanes 0-15 K=[0..7]+[16..23],
            // lanes 16-31 K=[8..15]+[24..31]; row = lane%16.
            v16bf af[4];
#pragma unroll
            for (int i = 0; i < 4; i++) {
                const __bf16* rp = &As[cur][(wm * 64 + i * 16 + lr) * LDSS + kk];
                v8bf lo = *(const v8bf*)&rp[half * 8];
                v8bf hi = *(const v8bf*)&rp[16 + half * 8];
                af[i] = mkfrag(lo, hi);
            }
            // B frag (16-bit B 32x16): lanes 0-15 col n K=[0..7]+[16..23], lanes 16-31 K=[8..15]+[24..31].
            v16bf bfm[2];
#pragma unroll
            for (int j = 0; j < 2; j++) {
                const __bf16* rp = &Bs[cur][(wn * 32 + j * 16 + lr) * LDSS + kk];
                v8bf lo = *(const v8bf*)&rp[half * 8];
                v8bf hi = *(const v8bf*)&rp[16 + half * 8];
                bfm[j] = mkfrag(lo, hi);
            }
#pragma unroll
            for (int i = 0; i < 4; i++)
#pragma unroll
                for (int j = 0; j < 2; j++)
                    acc[i][j] = __builtin_amdgcn_wmma_f32_16x16x32_bf16(
                        false, af[i], false, bfm[j], (short)0, acc[i][j], false, false);
        }

        wait_async_copies();
        __syncthreads();
    }

    // C layout: VGPR r -> (lanes 0-15: M=r, N=lane), (lanes 16-31: M=r+8, N=lane-16)
#pragma unroll
    for (int i = 0; i < 4; i++)
#pragma unroll
        for (int j = 0; j < 2; j++) {
            const int mrow = mBase + wm * 64 + i * 16 + half * 8;
            const int ncol = nBase + wn * 32 + j * 16 + lr;
#pragma unroll
            for (int r = 0; r < 8; r++)
                Cg[(size_t)(mrow + r) * ldc + ncol] = cvt_out<OUT_T>(acc[i][j][r]);
        }
}

// ---------------------------------------------------------------------------
// f32 -> bf16 elementwise (n multiple of 1024)
// ---------------------------------------------------------------------------
__global__ __launch_bounds__(256)
void cvt_f32_bf16(const float* __restrict__ src, __bf16* __restrict__ dst, int n)
{
    int i = (blockIdx.x * 256 + threadIdx.x) * 4;
    if (i < n) {
        float4 f = *(const float4*)(src + i);
        dst[i + 0] = f32_to_bf16(f.x);
        dst[i + 1] = f32_to_bf16(f.y);
        dst[i + 2] = f32_to_bf16(f.z);
        dst[i + 3] = f32_to_bf16(f.w);
    }
}

// ---------------------------------------------------------------------------
// scale + mask + softmax over rows of LK=1024. grid = (LQ, B*H), block = 256.
// ---------------------------------------------------------------------------
__global__ __launch_bounds__(256)
void softmax_mask_kernel(const __bf16* __restrict__ scores, const int* __restrict__ mask,
                         __bf16* __restrict__ attn, float scale)
{
    const int row = blockIdx.x;
    const int z   = blockIdx.y;       // b*H + h
    const int b   = z / Hn;
    const __bf16* srow = scores + ((size_t)z * LQn + row) * LKn;
    const int*    mrow = mask   + ((size_t)b * LQn + row) * LKn;
    __bf16*       arow = attn   + ((size_t)z * LQn + row) * LKn;

    const int t = threadIdx.x;
    const float NEG_INF = -__builtin_inff();

    float v[4];
#pragma unroll
    for (int i = 0; i < 4; i++) {
        int c = t + i * 256;
        float s = bf16_to_f32(srow[c]) * scale;
        v[i] = (mrow[c] == 0) ? NEG_INF : s;
    }

    __shared__ float red[256];
    float m = fmaxf(fmaxf(v[0], v[1]), fmaxf(v[2], v[3]));
    red[t] = m;
    __syncthreads();
    for (int s = 128; s > 0; s >>= 1) {
        if (t < s) red[t] = fmaxf(red[t], red[t + s]);
        __syncthreads();
    }
    m = red[0];
    __syncthreads();

    float e[4], sum = 0.0f;
#pragma unroll
    for (int i = 0; i < 4; i++) { e[i] = __expf(v[i] - m); sum += e[i]; }
    red[t] = sum;
    __syncthreads();
    for (int s = 128; s > 0; s >>= 1) {
        if (t < s) red[t] += red[t + s];
        __syncthreads();
    }
    float inv = 1.0f / red[0];
#pragma unroll
    for (int i = 0; i < 4; i++) arow[t + i * 256] = f32_to_bf16(e[i] * inv);
}

// ---------------------------------------------------------------------------
// host launcher
// ---------------------------------------------------------------------------
extern "C" void kernel_launch(void* const* d_in, const int* in_sizes, int n_in,
                              void* d_out, int out_size, void* d_ws, size_t ws_size,
                              hipStream_t stream)
{
    (void)in_sizes; (void)n_in; (void)out_size; (void)ws_size;

    const float* x      = (const float*)d_in[0];
    const float* states = (const float*)d_in[1];
    const int*   mask   = (const int*)  d_in[2];
    const float* Wq     = (const float*)d_in[3];
    const float* Wk     = (const float*)d_in[4];
    const float* Wv     = (const float*)d_in[5];
    const float* Wp     = (const float*)d_in[6];
    float*       out    = (float*)d_out;

    // workspace layout (bytes); lifetime-based aliasing:
    //   attn aliases [qb..vtb) (dead after scores GEMM)
    //   ctx  aliases scores   (dead after softmax)
    const size_t MB = 1u << 20;
    char* ws = (char*)d_ws;
    __bf16* xb     = (__bf16*)(ws + 0 * MB);    //  4 MB  [B,LQ,D]
    __bf16* sb     = (__bf16*)(ws + 4 * MB);    //  4 MB  [B,LK,D]
    __bf16* wqb    = (__bf16*)(ws + 8 * MB);    //  4 MB  [H,D,D]
    __bf16* wkb    = (__bf16*)(ws + 12 * MB);   //  4 MB
    __bf16* wvb    = (__bf16*)(ws + 16 * MB);   //  4 MB
    __bf16* wpb    = (__bf16*)(ws + 20 * MB);   //  4 MB  [D,H*D]
    __bf16* qb     = (__bf16*)(ws + 24 * MB);   // 32 MB  [B,H,LQ,D]
    __bf16* kb     = (__bf16*)(ws + 56 * MB);   // 32 MB  [B,H,LK,D]
    __bf16* vtb    = (__bf16*)(ws + 88 * MB);   // 32 MB  [B,H,D,LK]
    __bf16* scores = (__bf16*)(ws + 120 * MB);  // 64 MB  [B,H,LQ,LK]
    __bf16* attnb  = qb;                        // 64 MB alias over qb+kb
    __bf16* ctxb   = scores;                    // 32 MB alias over scores
    // total workspace footprint: 184 MB

    // phase 0: conversions (each 2,097,152 elements)
    const int nconv = 2097152;
    const int gcv   = nconv / (256 * 4);
    cvt_f32_bf16<<<gcv, 256, 0, stream>>>(x,      xb,  nconv);
    cvt_f32_bf16<<<gcv, 256, 0, stream>>>(states, sb,  nconv);
    cvt_f32_bf16<<<gcv, 256, 0, stream>>>(Wq,     wqb, nconv);
    cvt_f32_bf16<<<gcv, 256, 0, stream>>>(Wk,     wkb, nconv);
    cvt_f32_bf16<<<gcv, 256, 0, stream>>>(Wv,     wvb, nconv);
    cvt_f32_bf16<<<gcv, 256, 0, stream>>>(Wp,     wpb, nconv);

    const long long LQD  = (long long)LQn * Dn;
    const long long LKD  = (long long)LKn * Dn;
    const long long DDq  = (long long)Dn * Dn;
    const long long DLK  = (long long)Dn * LKn;
    const long long LQLK = (long long)LQn * LKn;
    const long long LQHD = (long long)LQn * HDn;

    // phase 1: q = x @ Wq[h]^T   (z = b*H + h)
    gemm_bf16_abt<__bf16><<<dim3(Dn / 128, LQn / 128, Bsz * Hn), 256, 0, stream>>>(
        xb, wqb, qb, LQn, Dn, Dn, Dn, Dn, Dn,
        BMap{LQD, 0, Hn}, BMap{0, DDq, Hn}, BMap{LQD, 0, 1});
    // phase 2: k = states @ Wk[h]^T
    gemm_bf16_abt<__bf16><<<dim3(Dn / 128, LKn / 128, Bsz * Hn), 256, 0, stream>>>(
        sb, wkb, kb, LKn, Dn, Dn, Dn, Dn, Dn,
        BMap{LKD, 0, Hn}, BMap{0, DDq, Hn}, BMap{LKD, 0, 1});
    // phase 3: vT = Wv[h] @ states^T  -> [D, LK]
    gemm_bf16_abt<__bf16><<<dim3(LKn / 128, Dn / 128, Bsz * Hn), 256, 0, stream>>>(
        wvb, sb, vtb, Dn, LKn, Dn, Dn, Dn, LKn,
        BMap{0, DDq, Hn}, BMap{LKD, 0, Hn}, BMap{DLK, 0, 1});
    // phase 4: scores = q @ k^T
    gemm_bf16_abt<__bf16><<<dim3(LKn / 128, LQn / 128, Bsz * Hn), 256, 0, stream>>>(
        qb, kb, scores, LQn, LKn, Dn, Dn, Dn, LKn,
        BMap{LQD, 0, 1}, BMap{LKD, 0, 1}, BMap{LQLK, 0, 1});
    // phase 5: attn = softmax(mask(scale * scores))
    const float scale = 1.0f / 22.627416997969522f;   // 1/sqrt(512)
    softmax_mask_kernel<<<dim3(LQn, Bsz * Hn), 256, 0, stream>>>(scores, mask, attnb, scale);
    // phase 6: ctx = attn @ v = attn @ vT^T, stored head-major [B, LQ, H*D] (ldc = 4096)
    gemm_bf16_abt<__bf16><<<dim3(Dn / 128, LQn / 128, Bsz * Hn), 256, 0, stream>>>(
        attnb, vtb, ctxb, LQn, Dn, LKn, LKn, LKn, HDn,
        BMap{LQLK, 0, 1}, BMap{DLK, 0, 1}, BMap{LQHD, (long long)Dn, Hn});
    // phase 7: out = ctx_cat @ Wp^T  (f32 output)
    gemm_bf16_abt<float><<<dim3(Dn / 128, LQn / 128, Bsz), 256, 0, stream>>>(
        ctxb, wpb, out, LQn, Dn, HDn, HDn, HDn, Dn,
        BMap{LQHD, 0, 1}, BMap{0, 0, 1}, BMap{LQD, 0, 1});
}
